// Plane_45105746542680
// MI455X (gfx1250) — matrix operand order repaired and checked by reading, not compiled, not run on hardware
//
#include <hip/hip_runtime.h>
#include <stdint.h>

// ---------------------------------------------------------------------------
// Plane forward: 4-corner bilinear blend of (hashgrid encode -> 32->64->8 MLP)
// WMMA f16->f32 for the MLP, scalar VALU + gathers for the hashgrid.
// ---------------------------------------------------------------------------

typedef _Float16 v16h __attribute__((ext_vector_type(16)));
typedef _Float16 v8h  __attribute__((ext_vector_type(8)));
typedef _Float16 h2   __attribute__((ext_vector_type(2)));
typedef float    v8f  __attribute__((ext_vector_type(8)));

// Per-level hashgrid constants (gridencoder: base 16 -> 2048, 16 levels, 2^19 cap)
static constexpr float    kScale[16] = {
    15.0f, 21.1106075f, 29.5549164f, 41.2242531f, 57.3502449f, 79.6349472f,
    110.4304705f, 152.9872449f, 211.7969003f, 293.0667788f, 405.3746693f,
    560.5743613f, 775.0468819f, 1071.4289946f, 1481.0037530f, 2047.0f};
static constexpr uint32_t kRes[16]  = {16,23,31,43,59,81,112,154,213,295,407,562,777,1073,1483,2048};
static constexpr uint32_t kSize[16] = {256,536,968,1856,3488,6568,12544,23720,45376,87032,
                                       165656,315848,524288,524288,524288,524288};
static constexpr uint32_t kOff[16]  = {0,256,792,1760,3616,7104,13672,26216,49936,95312,
                                       182344,348000,663848,1188136,1712424,2236712};
static constexpr uint32_t kPrimeY   = 2654435761u;

#define FEAT_STRIDE 40   // halves per feature row (32 data + pad -> 80B, 16B aligned, bank-skewed)
#define W1_STRIDE   72   // halves per w1/h row (64 data + pad -> 144B)

__global__ __launch_bounds__(128)
void Plane_45105746542680_kernel(const float* __restrict__ xy,
                                 const float* __restrict__ table,
                                 const float* __restrict__ w0,
                                 const float* __restrict__ w1,
                                 const int*   __restrict__ boundp,
                                 float*       __restrict__ out,
                                 int npoints)
{
    __shared__ alignas(16) _Float16 sFeat[128 * FEAT_STRIDE];  // A rows for layer 1
    __shared__ alignas(16) _Float16 sW0[64 * FEAT_STRIDE];     // B for layer 1: [n][k]
    __shared__ alignas(16) _Float16 sW1[16 * W1_STRIDE];       // B for layer 2: [o][k], rows 8..15 zero
    __shared__ alignas(16) _Float16 sH[4 * 16 * W1_STRIDE];    // per-wave relu(h1) in [m][k]
    __shared__ float sU[32], sV[32];

    const int t = threadIdx.x;

    // ---------------- stage weights (f32 -> f16) ----------------
    {
        const int n  = t >> 1;            // 0..63
        const int kb = (t & 1) * 16;      // 0 or 16
        #pragma unroll
        for (int j = 0; j < 16; ++j)
            sW0[n * FEAT_STRIDE + kb + j] = (_Float16)w0[n * 32 + kb + j];

        if (t < 64) {
            const int o   = t >> 3;       // 0..7
            const int kb2 = (t & 7) * 8;
            #pragma unroll
            for (int j = 0; j < 8; ++j)
                sW1[o * W1_STRIDE + kb2 + j] = (_Float16)w1[o * 64 + kb2 + j];
        } else {
            const int q   = t - 64;
            const int o   = 8 + (q >> 3); // 8..15 (zero pad rows)
            const int kb2 = (q & 7) * 8;
            #pragma unroll
            for (int j = 0; j < 8; ++j)
                sW1[o * W1_STRIDE + kb2 + j] = (_Float16)0.0f;
        }
    }

    // ---------------- hashgrid features: one (point,corner) row per thread ----
    {
        const int plocal = t >> 2;                       // 0..31
        const int c      = t & 3;                        // corner id
        const int p      = blockIdx.x * 32 + plocal;
        const int pc     = p < npoints ? p : (npoints - 1);

        const float2 xv = ((const float2*)xy)[pc];
        const float  bnd   = (float)boundp[0];
        const float  inv2b = 0.5f / bnd;

        const float xr = (xv.x + bnd) * inv2b;
        const float yr = (xv.y + bnd) * inv2b;
        const float cxx = fminf(fmaxf(xr * 2048.0f - 0.5f, 0.0f), 2047.0f);
        const float cyy = fminf(fmaxf(yr * 2048.0f - 0.5f, 0.0f), 2047.0f);
        const float cx0 = floorf(cxx), cy0 = floorf(cyy);
        const float u   = cxx - cx0,   v   = cyy - cy0;
        const float cx1 = fminf(cx0 + 1.0f, 2047.0f);
        const float cy1 = fminf(cy0 + 1.0f, 2047.0f);
        if (c == 0) { sU[plocal] = u; sV[plocal] = v; }

        const float ax = (c & 1) ? cx1 : cx0;
        const float ay = (c & 2) ? cy1 : cy0;
        const float gx = (ax + 0.5f) * (1.0f / 2048.0f);
        const float gy = (ay + 0.5f) * (1.0f / 2048.0f);

        const float2* tab = (const float2*)table;

        #pragma unroll
        for (int l = 0; l < 16; ++l) {
            const float px = fmaf(gx, kScale[l], 0.5f);
            const float py = fmaf(gy, kScale[l], 0.5f);
            const float fx = floorf(px), fy = floorf(py);
            const float frx = px - fx,   fry = py - fy;
            const uint32_t pgx = (uint32_t)fx, pgy = (uint32_t)fy;
            const float wx0 = 1.0f - frx, wy0 = 1.0f - fry;

            float a0 = 0.0f, a1 = 0.0f;
            #pragma unroll
            for (int dy = 0; dy < 2; ++dy) {
                #pragma unroll
                for (int dx = 0; dx < 2; ++dx) {
                    const uint32_t gxi = pgx + (uint32_t)dx;
                    const uint32_t gyi = pgy + (uint32_t)dy;
                    uint32_t idx;
                    if (l >= 12) {                       // hashed level, size = 2^19
                        idx = (gxi ^ (gyi * kPrimeY)) & (kSize[l] - 1u);
                    } else {                             // dense level, idx < 2*size
                        idx = gxi + gyi * kRes[l];
                        if (idx >= kSize[l]) idx -= kSize[l];
                    }
                    const float2 tv = tab[kOff[l] + idx];
                    const float  w  = (dx ? frx : wx0) * (dy ? fry : wy0);
                    a0 = fmaf(w, tv.x, a0);
                    a1 = fmaf(w, tv.y, a1);
                }
            }
            h2 ph; ph[0] = (_Float16)a0; ph[1] = (_Float16)a1;
            *(h2*)&sFeat[t * FEAT_STRIDE + 2 * l] = ph;
        }
    }

    __syncthreads();

    // ---------------- MLP via WMMA ----------------
    const int lane = t & 31;
    const int wv   = t >> 5;
    const int nrow = lane & 15;   // M for A, N for B/C/D
    const int khi  = lane >> 4;

    // Constant trip count so the compiler fully unrolls instead of building a
    // divergence-handled loop around the WMMAs (EXEC stays trivially all-ones).
    #pragma unroll
    for (int it = 0; it < 2; ++it) {
        const int mt = wv + it * 4;           // wave handles M-tiles wv and wv+4

        // --- layer-1 A fragment: ISA f16 A layout (K = khi*8+0..7 and 16+khi*8+0..7)
        const _Float16* fr = &sFeat[(mt * 16 + nrow) * FEAT_STRIDE];
        const v8h alo = *(const v8h*)(fr + khi * 8);
        const v8h ahi = *(const v8h*)(fr + 16 + khi * 8);
        const v16h a1 = __builtin_shufflevector(alo, ahi, 0,1,2,3,4,5,6,7,8,9,10,11,12,13,14,15);

        _Float16* hb = &sH[wv * 16 * W1_STRIDE];

        #pragma unroll
        for (int nt = 0; nt < 4; ++nt) {
            // --- layer-1 B fragment: same K order as A (K = khi*8+0..7 and 16+khi*8+0..7)
            const _Float16* wr = &sW0[(nt * 16 + nrow) * FEAT_STRIDE + khi * 8];
            const v8h blo = *(const v8h*)(wr);
            const v8h bhi = *(const v8h*)(wr + 16);
            const v16h b1 = __builtin_shufflevector(blo, bhi, 0,1,2,3,4,5,6,7,8,9,10,11,12,13,14,15);

            v8f d = {0.f,0.f,0.f,0.f,0.f,0.f,0.f,0.f};
            d = __builtin_amdgcn_wmma_f32_16x16x32_f16(false, a1, false, b1,
                                                       (short)0, d, false, false);
            // relu -> f16 -> LDS in [m][k] layout (k = hidden index)
            #pragma unroll
            for (int i = 0; i < 8; ++i) {
                const float r = d[i] > 0.0f ? d[i] : 0.0f;
                hb[(khi * 8 + i) * W1_STRIDE + nt * 16 + nrow] = (_Float16)r;
            }
        }

        __syncthreads();   // uniform (unrolled): hb visibility + safe reuse

        // --- layer-2 A fragments (K = 0..31 and 32..63 of hidden dim)
        const _Float16* hr = &sH[wv * 16 * W1_STRIDE + nrow * W1_STRIDE];
        const v8h x0 = *(const v8h*)(hr + khi * 8);
        const v8h x1 = *(const v8h*)(hr + 16 + khi * 8);
        const v8h x2 = *(const v8h*)(hr + 32 + khi * 8);
        const v8h x3 = *(const v8h*)(hr + 48 + khi * 8);
        const v16h a2a = __builtin_shufflevector(x0, x1, 0,1,2,3,4,5,6,7,8,9,10,11,12,13,14,15);
        const v16h a2b = __builtin_shufflevector(x2, x3, 0,1,2,3,4,5,6,7,8,9,10,11,12,13,14,15);

        // --- layer-2 B fragments
        const _Float16* w1r = &sW1[nrow * W1_STRIDE + khi * 8];
        const v8h y0 = *(const v8h*)(w1r);
        const v8h y1 = *(const v8h*)(w1r + 16);
        const v8h y2 = *(const v8h*)(w1r + 32);
        const v8h y3 = *(const v8h*)(w1r + 48);
        const v16h b2a = __builtin_shufflevector(y0, y1, 0,1,2,3,4,5,6,7,8,9,10,11,12,13,14,15);
        const v16h b2b = __builtin_shufflevector(y2, y3, 0,1,2,3,4,5,6,7,8,9,10,11,12,13,14,15);

        v8f d2 = {0.f,0.f,0.f,0.f,0.f,0.f,0.f,0.f};
        d2 = __builtin_amdgcn_wmma_f32_16x16x32_f16(false, a2a, false, b2a,
                                                    (short)0, d2, false, false);
        d2 = __builtin_amdgcn_wmma_f32_16x16x32_f16(false, a2b, false, b2b,
                                                    (short)0, d2, false, false);

        // --- bilinear blend of the 4 corner rows, write [point][nrow]
        const int lp0 = mt * 4 + khi * 2;          // local points held by this lane
        const float u0 = sU[lp0],     vv0 = sV[lp0];
        const float u1 = sU[lp0 + 1], vv1 = sV[lp0 + 1];

        const float o0 = (1.0f - u0) * (1.0f - vv0) * d2[0] + u0 * (1.0f - vv0) * d2[1]
                       + (1.0f - u0) * vv0          * d2[2] + u0 * vv0          * d2[3];
        const float o1 = (1.0f - u1) * (1.0f - vv1) * d2[4] + u1 * (1.0f - vv1) * d2[5]
                       + (1.0f - u1) * vv1          * d2[6] + u1 * vv1          * d2[7];

        const long long g0 = (long long)blockIdx.x * 32 + lp0;
        if (nrow < 8) {
            if (g0 < npoints)     out[g0 * 8 + nrow]       = o0;
            if (g0 + 1 < npoints) out[(g0 + 1) * 8 + nrow] = o1;
        }
    }
}

extern "C" void kernel_launch(void* const* d_in, const int* in_sizes, int n_in,
                              void* d_out, int out_size, void* d_ws, size_t ws_size,
                              hipStream_t stream) {
    const float* xy    = (const float*)d_in[0];
    const float* table = (const float*)d_in[1];
    const float* w0    = (const float*)d_in[2];
    const float* w1    = (const float*)d_in[3];
    const int*   bound = (const int*)  d_in[4];
    float*       out   = (float*)d_out;

    const int npoints = in_sizes[0] / 2;           // xy is [N,2]
    const int blocks  = (npoints + 31) / 32;       // 32 points (128 rows) per block

    Plane_45105746542680_kernel<<<dim3(blocks), dim3(128), 0, stream>>>(
        xy, table, w0, w1, bound, out, npoints);
}
